// render_4234837754028
// MI455X (gfx1250) — matrix operand order repaired and checked by reading, not compiled, not run on hardware
//
#include <hip/hip_runtime.h>
#include <hip/hip_bf16.h>
#include <math.h>

typedef __attribute__((ext_vector_type(16))) _Float16 v16h;
typedef __attribute__((ext_vector_type(8)))  float    v8f;
typedef __attribute__((ext_vector_type(4)))  unsigned int v4u;
typedef __attribute__((ext_vector_type(8)))  int      v8i;
typedef __attribute__((ext_vector_type(4)))  int      v4i;

#define NG    1024
#define IMGW  192
#define IMGH  192
#define NPIX  (IMGW*IMGH)
#define TANFOVF 0.5f

#if defined(__has_builtin)
#if __has_builtin(__builtin_amdgcn_tensor_load_to_lds) && __has_builtin(__builtin_amdgcn_s_wait_tensorcnt)
#define HAVE_TDM 1
#endif
#endif

// ---------------------------------------------------------------------------
// Kernel 1: per-gaussian preprocessing.
// Writes: params[g*8]  = {gx, gy, conA, conB, conC, opacity, radius, visible}
//         colw[g*4]    = {r, g, b, depthKey (inf if invisible)}
//         d_out radii / visibility / means2D sections.
// ---------------------------------------------------------------------------
__global__ void __launch_bounds__(256) gs_preprocess(
    const float* __restrict__ means3D, const float* __restrict__ scales,
    const float* __restrict__ rotations, const float* __restrict__ opacities,
    const float* __restrict__ shs, const float* __restrict__ viewm,
    const float* __restrict__ projm, const float* __restrict__ campos,
    float* __restrict__ params, float* __restrict__ colw,
    float* __restrict__ outF)
{
    int g = blockIdx.x * blockDim.x + threadIdx.x;
    if (g >= NG) return;

    float mx = means3D[g*3+0], my = means3D[g*3+1], mz = means3D[g*3+2];

    // view direction
    float dx = mx - campos[0], dy = my - campos[1], dz = mz - campos[2];
    float rinv = rsqrtf(dx*dx + dy*dy + dz*dz);
    float x = dx*rinv, y = dy*rinv, z = dz*rinv;
    float xx = x*x, yy = y*y, zz = z*z, xy = x*y, yz = y*z, xz = x*z;

    // SH basis (degree 3)
    float bas[16];
    bas[0]  =  0.28209479177387814f;
    bas[1]  = -0.4886025119029199f * y;
    bas[2]  =  0.4886025119029199f * z;
    bas[3]  = -0.4886025119029199f * x;
    bas[4]  =  1.0925484305920792f * xy;
    bas[5]  = -1.0925484305920792f * yz;
    bas[6]  =  0.31539156525252005f * (2.f*zz - xx - yy);
    bas[7]  = -1.0925484305920792f * xz;
    bas[8]  =  0.5462742152960396f * (xx - yy);
    bas[9]  = -0.5900435899266435f * y * (3.f*xx - yy);
    bas[10] =  2.890611442640554f  * xy * z;
    bas[11] = -0.4570457994644658f * y * (4.f*zz - xx - yy);
    bas[12] =  0.3731763325901154f * z * (2.f*zz - 3.f*xx - 3.f*yy);
    bas[13] = -0.4570457994644658f * x * (4.f*zz - xx - yy);
    bas[14] =  1.445305721320277f  * z * (xx - yy);
    bas[15] = -0.5900435899266435f * x * (xx - yy - 3.f*zz);

    float col[3];
    #pragma unroll
    for (int c = 0; c < 3; ++c) {
        float s = 0.f;
        #pragma unroll
        for (int i = 0; i < 16; ++i) s += bas[i] * shs[g*48 + i*3 + c];
        col[c] = fmaxf(s + 0.5f, 0.f);
    }

    // quaternion -> rotation, cov3d = (R*S)(R*S)^T
    float qr = rotations[g*4+0], qx = rotations[g*4+1];
    float qy = rotations[g*4+2], qz = rotations[g*4+3];
    float R00 = 1.f-2.f*(qy*qy+qz*qz), R01 = 2.f*(qx*qy-qr*qz), R02 = 2.f*(qx*qz+qr*qy);
    float R10 = 2.f*(qx*qy+qr*qz), R11 = 1.f-2.f*(qx*qx+qz*qz), R12 = 2.f*(qy*qz-qr*qx);
    float R20 = 2.f*(qx*qz-qr*qy), R21 = 2.f*(qy*qz+qr*qx), R22 = 1.f-2.f*(qx*qx+qy*qy);
    float s0 = scales[g*3+0], s1 = scales[g*3+1], s2 = scales[g*3+2];
    float M00 = R00*s0, M01 = R01*s1, M02 = R02*s2;
    float M10 = R10*s0, M11 = R11*s1, M12 = R12*s2;
    float M20 = R20*s0, M21 = R21*s1, M22 = R22*s2;
    float cov[3][3];
    cov[0][0] = M00*M00 + M01*M01 + M02*M02;
    cov[0][1] = M00*M10 + M01*M11 + M02*M12;
    cov[0][2] = M00*M20 + M01*M21 + M02*M22;
    cov[1][1] = M10*M10 + M11*M11 + M12*M12;
    cov[1][2] = M10*M20 + M11*M21 + M12*M22;
    cov[2][2] = M20*M20 + M21*M21 + M22*M22;
    cov[1][0] = cov[0][1]; cov[2][0] = cov[0][2]; cov[2][1] = cov[1][2];

    // view & projection transforms (hom = [m,1])
    float pv[3], ph[4];
    #pragma unroll
    for (int i = 0; i < 3; ++i)
        pv[i] = viewm[i*4+0]*mx + viewm[i*4+1]*my + viewm[i*4+2]*mz + viewm[i*4+3];
    #pragma unroll
    for (int i = 0; i < 4; ++i)
        ph[i] = projm[i*4+0]*mx + projm[i*4+1]*my + projm[i*4+2]*mz + projm[i*4+3];

    float invw = 1.f / (ph[3] + 1e-7f);
    float gx = ((ph[0]*invw + 1.f) * (float)IMGW - 1.f) * 0.5f;
    float gy = ((ph[1]*invw + 1.f) * (float)IMGH - 1.f) * 0.5f;
    float depth = pv[2];

    bool in_front = depth > 0.2f;
    float tz = in_front ? depth : 1.f;
    const float lim = 1.3f * TANFOVF;
    float tx = fminf(fmaxf(pv[0]/tz, -lim), lim) * tz;
    float ty = fminf(fmaxf(pv[1]/tz, -lim), lim) * tz;
    const float fx = (float)IMGW / (2.f*TANFOVF);
    const float fy = (float)IMGH / (2.f*TANFOVF);
    float itz = 1.f/tz, itz2 = itz*itz;
    float J00 = fx*itz, J02 = -fx*tx*itz2;
    float J11 = fy*itz, J12 = -fy*ty*itz2;

    // T2 = J @ V[:3,:3]
    float T0[3], T1[3];
    #pragma unroll
    for (int k = 0; k < 3; ++k) {
        T0[k] = J00*viewm[0*4+k] + J02*viewm[2*4+k];
        T1[k] = J11*viewm[1*4+k] + J12*viewm[2*4+k];
    }
    // cov2d = T2 cov3d T2^T
    float t0[3], t1[3];
    #pragma unroll
    for (int k = 0; k < 3; ++k) {
        t0[k] = T0[0]*cov[0][k] + T0[1]*cov[1][k] + T0[2]*cov[2][k];
        t1[k] = T1[0]*cov[0][k] + T1[1]*cov[1][k] + T1[2]*cov[2][k];
    }
    float A = t0[0]*T0[0] + t0[1]*T0[1] + t0[2]*T0[2] + 0.3f;
    float B = t0[0]*T1[0] + t0[1]*T1[1] + t0[2]*T1[2];
    float C = t1[0]*T1[0] + t1[1]*T1[1] + t1[2]*T1[2] + 0.3f;

    float det = A*C - B*B;
    bool posd = det > 0.f;
    float invdet = posd ? 1.f/det : 0.f;
    float conA = C*invdet, conB = -B*invdet, conC = A*invdet;
    float mid = 0.5f*(A + C);
    float lam = mid + sqrtf(fmaxf(mid*mid - det, 0.1f));
    bool visible = in_front && posd;
    float radius = visible ? ceilf(3.f * sqrtf(lam)) : 0.f;

    params[g*8+0] = gx;   params[g*8+1] = gy;
    params[g*8+2] = conA; params[g*8+3] = conB;
    params[g*8+4] = conC; params[g*8+5] = opacities[g];
    params[g*8+6] = radius; params[g*8+7] = visible ? 1.f : 0.f;

    colw[g*4+0] = col[0]; colw[g*4+1] = col[1]; colw[g*4+2] = col[2];
    colw[g*4+3] = visible ? depth : 3.0e38f;

    int* radOut = (int*)(outF + 3*NPIX);
    radOut[g] = (int)radius;
    int* visOut = (int*)(outF + 3*NPIX + NG);
    visOut[g] = (radius > 0.f) ? 1 : 0;
    float* m2 = outF + 3*NPIX + 2*NG;
    m2[g*2+0] = gx; m2[g*2+1] = gy;
}

// ---------------------------------------------------------------------------
// Kernel 2: single-block bitonic sort by depth key + gather sorted params and
// build WMMA B fragments (32 chunks x 32 lanes x 16 halves, per-lane layout).
// ---------------------------------------------------------------------------
__global__ void __launch_bounds__(256) gs_sort_gather(
    const float* __restrict__ params, const float* __restrict__ colw,
    float* __restrict__ sortedParams, _Float16* __restrict__ Bfrag)
{
    __shared__ float key[NG];
    __shared__ int   id[NG];
    const int tid = threadIdx.x;

    for (int i = tid; i < NG; i += 256) { key[i] = colw[i*4+3]; id[i] = i; }
    __syncthreads();

    for (int k = 2; k <= NG; k <<= 1) {
        for (int j = k >> 1; j > 0; j >>= 1) {
            for (int i = tid; i < NG; i += 256) {
                int ixj = i ^ j;
                if (ixj > i) {
                    bool up = ((i & k) == 0);
                    float ki = key[i], kj = key[ixj];
                    bool sw = up ? (ki > kj) : (ki < kj);
                    if (sw) {
                        key[i] = kj; key[ixj] = ki;
                        int t = id[i]; id[i] = id[ixj]; id[ixj] = t;
                    }
                }
            }
            __syncthreads();
        }
    }

    // gather params in sorted order
    for (int g = tid; g < NG; g += 256) {
        int o = id[g];
        #pragma unroll
        for (int c = 0; c < 8; ++c) sortedParams[g*8+c] = params[o*8+c];
    }

    // build B fragments: entry e = chunk*32 + lane; lanes 0-15 carry K=0..15
    // of the chunk for column N=lane; lanes 16-31 carry K=16..31 for N=lane-16.
    for (int e = tid; e < NG; e += 256) {
        int chunk = e >> 5, lane = e & 31;
        int col   = lane & 15;
        int kbase = (lane < 16) ? 0 : 16;
        #pragma unroll
        for (int j = 0; j < 16; ++j) {
            int gI = chunk*32 + kbase + j;
            int o = id[gI];
            float val = 0.f;
            if (col < 3) val = colw[o*4+col];
            Bfrag[e*16 + j] = (_Float16)val;
        }
    }
}

// ---------------------------------------------------------------------------
// Kernel 3: rasterization. Each wave owns 16 pixels (lane pairs m / m+16 both
// run the serial transmittance scan for pixel m, packing complementary halves
// of the WMMA A fragment). Colors enter through precomputed B fragments; the
// weighted color accumulation is v_wmma_f32_16x16x32_f16. Gaussian params are
// staged global->LDS via the Tensor Data Mover (one 1-D D# issued by wave 0,
// tracked with TENSORcnt), falling back to cooperative copies if the builtin
// is unavailable.
// ---------------------------------------------------------------------------
__global__ void __launch_bounds__(256) gs_raster(
    const float* __restrict__ sortedParams, const _Float16* __restrict__ Bfrag,
    const float* __restrict__ bg, float* __restrict__ out)
{
    __shared__ float sp[NG*8];   // 32 KB of sorted gaussian params
    const int tid = threadIdx.x;

#if defined(HAVE_TDM)
    if (tid < 32) {
        // Tensor DMA descriptor: 1-D tensor of 8192 dwords, tile = whole tensor.
        unsigned long long ga = (unsigned long long)sortedParams;
        unsigned int lds = (unsigned int)(unsigned long long)(void*)sp; // addr[31:0] = LDS offset
        v4u g0;
        g0[0] = 1u;                                   // count=1 (valid user D#)
        g0[1] = lds;                                  // lds_addr (bytes)
        g0[2] = (unsigned int)(ga & 0xFFFFFFFFu);     // global_addr[31:0]
        g0[3] = (unsigned int)((ga >> 32) & 0x01FFFFFFu) | (2u << 30); // addr[56:32] | type=2
        v8i g1;
        g1[0] = (int)(2u << 16);        // wg_mask=0, data_size=2 (4 bytes)
        g1[1] = (int)(8192u << 16);     // tensor_dim0[15:0]=8192 (elements)
        g1[2] = (int)(1u << 16);        // tensor_dim0 hi=0, tensor_dim1=1
        g1[3] = (int)(8192u << 16);     // tensor_dim1 hi=0, tile_dim0=8192
        g1[4] = 1;                      // tile_dim1=1, tile_dim2=0
        g1[5] = 8192;                   // tensor_dim0_stride[31:0]
        g1[6] = 0;                      // stride hi / tensor_dim1_stride lo
        g1[7] = 0;
        v4i z4 = {0, 0, 0, 0};
        v8i z8 = {0, 0, 0, 0, 0, 0, 0, 0};
        __builtin_amdgcn_tensor_load_to_lds(g0, g1, z4, z4, z8, 0);
        __builtin_amdgcn_s_wait_tensorcnt(0);
    }
    __syncthreads();
#else
    {
        const float4* gp4 = (const float4*)sortedParams;
        float4* sp4w = (float4*)sp;
        for (int i = tid; i < NG*2; i += 256) sp4w[i] = gp4[i];
    }
    __syncthreads();
#endif

    const int lane = tid & 31;
    const int wave = tid >> 5;
    const int tileBase = blockIdx.x * 128 + wave * 16;   // 288 blocks * 128 px
    const int m16 = lane & 15;
    const int pix = tileBase + m16;
    const float pxf = (float)(pix % IMGW);
    const float pyf = (float)(pix / IMGW);
    const bool lo = lane < 16;

    float T = 1.f;
    v8f acc = {};
    const float4* sp4 = (const float4*)sp;

    for (int chunk = 0; chunk < NG/32; ++chunk) {
        if (chunk + 1 < NG/32)
            __builtin_prefetch(Bfrag + (size_t)(chunk+1)*512 + (size_t)lane*16, 0, 1);

        float w32[32];
        #pragma unroll
        for (int kk = 0; kk < 32; ++kk) {
            int gI = chunk*32 + kk;
            float4 P0 = sp4[gI*2+0];   // gx, gy, conA, conB
            float4 P1 = sp4[gI*2+1];   // conC, opacity, radius, visible
            float ddx = P0.x - pxf;
            float ddy = P0.y - pyf;
            float power = -0.5f*(P0.z*ddx*ddx + P1.x*ddy*ddy) - P0.w*ddx*ddy;
            bool keep = (fabsf(ddx) <= P1.z) && (fabsf(ddy) <= P1.z) &&
                        (P1.w > 0.5f) && (power <= 0.f);
            float alpha = fminf(0.99f, P1.y * __expf(fminf(power, 0.f)));
            alpha = (keep && (alpha >= (1.f/255.f))) ? alpha : 0.f;
            w32[kk] = alpha * T;
            T *= (1.f - alpha);
        }

        // pack A fragment in contiguous K order (matches Bfrag):
        // lanes 0-15 (M=lane): a[j] = K j
        // lanes 16-31 (M=lane-16): a[j] = K j+16
        v16h a;
        #pragma unroll
        for (int j = 0; j < 16; ++j) {
            const int klo = j;
            const int khi = j + 16;
            a[j] = (_Float16)(lo ? w32[klo] : w32[khi]);
        }

        v16h b = *(const v16h*)(Bfrag + (size_t)(chunk*32 + lane)*16);

        acc = __builtin_amdgcn_wmma_f32_16x16x32_f16(
            /*neg_a=*/false, a, /*neg_b=*/false, b,
            /*c_mod=*/(short)0, acc, /*reuse_a=*/false, /*reuse_b=*/false);
    }

    // D layout: lanes 0-15 (col=lane) hold rows 0..7 in acc[0..7];
    //           lanes 16-31 (col=lane-16) hold rows 8..15.
    int col = lo ? lane : lane - 16;
    if (col < 3) {
        int rbase = lo ? 0 : 8;
        #pragma unroll
        for (int m = 0; m < 8; ++m) {
            float t = __shfl(T, rbase + m, 32);   // final transmittance of row's pixel
            int p = tileBase + rbase + m;
            out[col*NPIX + p] = acc[m] + t * bg[col];
        }
    }
}

// ---------------------------------------------------------------------------
extern "C" void kernel_launch(void* const* d_in, const int* in_sizes, int n_in,
                              void* d_out, int out_size, void* d_ws, size_t ws_size,
                              hipStream_t stream)
{
    const float* means3D = (const float*)d_in[0];
    const float* scales  = (const float*)d_in[1];
    const float* rot     = (const float*)d_in[2];
    const float* opac    = (const float*)d_in[3];
    const float* shs     = (const float*)d_in[4];
    const float* viewm   = (const float*)d_in[5];
    const float* projm   = (const float*)d_in[6];
    const float* campos  = (const float*)d_in[7];
    const float* bg      = (const float*)d_in[8];

    float* outF = (float*)d_out;
    float* wsf = (float*)d_ws;
    float* params       = wsf;            // 1024*8 floats
    float* colw         = wsf + 8192;     // 1024*4 floats (rgb + depth key)
    float* sortedParams = wsf + 12288;    // 1024*8 floats
    _Float16* Bfrag     = (_Float16*)(wsf + 20480); // 32*32*16 halves (32 KB)

    gs_preprocess<<<4, 256, 0, stream>>>(means3D, scales, rot, opac, shs,
                                         viewm, projm, campos,
                                         params, colw, outF);
    gs_sort_gather<<<1, 256, 0, stream>>>(params, colw, sortedParams, Bfrag);
    gs_raster<<<288, 256, 0, stream>>>(sortedParams, Bfrag, bg, outF);
}
